// KernelizedLinformerAttention_3865470567172
// MI455X (gfx1250) — matrix operand order repaired and checked by reading, not compiled, not run on hardware
//
#include <hip/hip_runtime.h>

typedef __attribute__((ext_vector_type(16))) _Float16 v16h;
typedef __attribute__((ext_vector_type(8)))  _Float16 v8h;
typedef __attribute__((ext_vector_type(8)))  float    v8f;

// Problem constants
// B=128, S=100, ED=1024, H=16, L=48, D=64, M=64
// BS = B*S = 12800, BH = B*H = 2048

__device__ __forceinline__ v8f wmma16(v16h a, v16h b, v8f c) {
  // D = A(16x32 f16) * B(32x16 f16) + C(16x16 f32)
  return __builtin_amdgcn_wmma_f32_16x16x32_f16(
      /*neg_a=*/false, a, /*neg_b=*/false, b,
      /*c_mod=*/(short)0, c, /*reuse_a=*/false, /*reuse_b=*/false);
}

// A fragment: 16x32 f16, row-major src with leading dim lda.
// Lane layout: row = lane&15; lanes0-15 hold K 0..15,
// lanes16-31 hold K 16..31.
__device__ __forceinline__ v16h load_a_frag(const _Float16* __restrict__ base,
                                            int lda, int row, int k0, int hsel) {
  const _Float16* p = base + (size_t)row * lda + k0 + hsel * 16;
  return *(const v16h*)p;
}

// B fragment from LDS tile stored n-major with leading dim ldb (k contiguous).
// Lane layout: col = lane&15; lanes0-15 K=0..15, lanes16-31 K=16..31.
__device__ __forceinline__ v16h load_b_frag_lds(const _Float16* base, int ldb,
                                                int col, int k0, int hsel) {
  return *(const v16h*)(base + (size_t)col * ldb + k0 + hsel * 16);
}

// ---------------- async global -> LDS copy (CDNA5 ASYNCcnt path) ------------
__device__ __forceinline__ void async_ld16(const void* gsrc, void* ldst) {
  // vdst = 32-bit LDS offset (low 32 bits of the generic LDS address),
  // vaddr = 64-bit global address, no saddr.
  unsigned lds_off = (unsigned)(unsigned long long)ldst;
  asm volatile("global_load_async_to_lds_b128 %0, %1, off"
               :: "v"(lds_off), "v"(gsrc)
               : "memory");
}
__device__ __forceinline__ void wait_async0() {
  asm volatile("s_wait_asynccnt 0x0" ::: "memory");
}

// ---------------- conversions ----------------
__global__ void k_f32tof16(const float* __restrict__ src,
                           _Float16* __restrict__ dst, int n) {
  int i = blockIdx.x * blockDim.x + threadIdx.x;
  if (i < n) dst[i] = (_Float16)src[i];
}

// ---------------- QKV GEMM: [12800,1024] x W[z]^T -> q/k (f16) or v (f32) ----
// B tile (64 cols x 32 k = 4KB) double-buffered in LDS via async copies.
__global__ __launch_bounds__(256) void k_gemm_qkv(
    const _Float16* __restrict__ xh, const _Float16* __restrict__ Wh,
    const float* __restrict__ bq, const float* __restrict__ bk,
    const float* __restrict__ bvv,
    _Float16* __restrict__ qh, _Float16* __restrict__ kh,
    float* __restrict__ vout, float* __restrict__ partials) {
  __shared__ alignas(128) _Float16 sB[2][64 * 32];  // 2 x 4KB
  const int z = blockIdx.z;
  const int wave = threadIdx.x >> 5;
  const int lane = threadIdx.x & 31;
  const int hsel = lane >> 4;
  const int l15  = lane & 15;
  const int row0 = blockIdx.x * 128 + wave * 16;
  const int col0 = blockIdx.y * 64;
  const _Float16* W = Wh + (size_t)z * (1024 * 1024);

  // each of 256 threads copies one 16B chunk of the 4KB tile
  const int cn  = threadIdx.x >> 2;        // column 0..63
  const int ck8 = (threadIdx.x & 3) * 8;   // k sub-offset 0,8,16,24

  const v8f vzero = {0.f,0.f,0.f,0.f,0.f,0.f,0.f,0.f};
  v8f acc[4];
#pragma unroll
  for (int i = 0; i < 4; ++i) acc[i] = vzero;

  // prefetch first B tile
  async_ld16(W + (size_t)(col0 + cn) * 1024 + 0 + ck8, &sB[0][cn * 32 + ck8]);
  wait_async0();
  __syncthreads();

  const int rowA = row0 + l15;
  int buf = 0;
  for (int k0 = 0; k0 < 1024; k0 += 32) {
    if (k0 + 32 < 1024)
      async_ld16(W + (size_t)(col0 + cn) * 1024 + (k0 + 32) + ck8,
                 &sB[buf ^ 1][cn * 32 + ck8]);
    v16h a = load_a_frag(xh, 1024, rowA, k0, hsel);
#pragma unroll
    for (int nt = 0; nt < 4; ++nt) {
      v16h bf = load_b_frag_lds(sB[buf], 32, nt * 16 + l15, 0, hsel);
      acc[nt] = wmma16(a, bf, acc[nt]);
    }
    wait_async0();
    __syncthreads();
    buf ^= 1;
  }

  const float* bias = (z == 0) ? bq : (z == 1) ? bk : bvv;
  float ss = 0.f;
#pragma unroll
  for (int nt = 0; nt < 4; ++nt) {
    const int col = col0 + nt * 16 + l15;
    const float bval = bias[col];
#pragma unroll
    for (int r = 0; r < 8; ++r) {
      const int row = row0 + r + hsel * 8;
      const float val = acc[nt][r] + bval;
      if (z == 2) {
        vout[(size_t)row * 1024 + col] = val;
      } else {
        (z == 0 ? qh : kh)[(size_t)row * 1024 + col] = (_Float16)val;
        ss += val * val;
      }
    }
  }
  if (z < 2) {
#pragma unroll
    for (int off = 16; off > 0; off >>= 1) ss += __shfl_xor(ss, off, 32);
    if (lane == 0)
      partials[z * 12800 + (blockIdx.x * 16 + blockIdx.y) * 8 + wave] = ss;
  }
}

// ---------------- deterministic norm reduction -> 1/||q||, 1/||k|| ----------
__global__ __launch_bounds__(256) void k_norm(const float* __restrict__ partials,
                                              float* __restrict__ inv) {
  __shared__ float red[256];
  const int z = blockIdx.x;
  float s = 0.f;
  for (int i = threadIdx.x; i < 12800; i += 256) s += partials[z * 12800 + i];
  red[threadIdx.x] = s;
  __syncthreads();
  for (int stride = 128; stride > 0; stride >>= 1) {
    if ((int)threadIdx.x < stride) red[threadIdx.x] += red[threadIdx.x + stride];
    __syncthreads();
  }
  if (threadIdx.x == 0) inv[z] = rsqrtf(red[0]);
}

// ---------------- phi = exp(inv * (q @ Wrf_h^T) - 0.5) / 64 -----------------
// Wrf[h] (64x64 f16 = 8KB) staged once per block in LDS via async copies.
__global__ __launch_bounds__(256) void k_phi(
    const _Float16* __restrict__ qh, const _Float16* __restrict__ kh,
    const _Float16* __restrict__ Wrfh, const float* __restrict__ inv,
    _Float16* __restrict__ phi_q16, _Float16* __restrict__ phi_k16) {
  __shared__ alignas(128) _Float16 sW[64 * 64];  // 8KB
  const int which = blockIdx.z;
  const _Float16* src = which ? kh : qh;
  _Float16* dst = which ? phi_k16 : phi_q16;
  const float scale = inv[which];
  const int h = blockIdx.y;
  const int wave = threadIdx.x >> 5;
  const int lane = threadIdx.x & 31;
  const int hsel = lane >> 4;
  const int l15  = lane & 15;
  const int row0 = blockIdx.x * 128 + wave * 16;

  const _Float16* Bp = Wrfh + h * 4096;  // Wrf[h][m][d], d contiguous
  for (int i = threadIdx.x; i < 512; i += 256)  // 512 x 16B = 8KB
    async_ld16(Bp + i * 8, &sW[i * 8]);
  wait_async0();
  __syncthreads();

  const v8f vzero = {0.f,0.f,0.f,0.f,0.f,0.f,0.f,0.f};
  v8f acc[4];
#pragma unroll
  for (int i = 0; i < 4; ++i) acc[i] = vzero;

  const _Float16* A = src + h * 64;  // row i stride 1024, d contiguous
#pragma unroll
  for (int k0 = 0; k0 < 64; k0 += 32) {
    v16h a = load_a_frag(A, 1024, row0 + l15, k0, hsel);
#pragma unroll
    for (int nt = 0; nt < 4; ++nt) {
      v16h bf = load_b_frag_lds(sW, 64, nt * 16 + l15, k0, hsel);
      acc[nt] = wmma16(a, bf, acc[nt]);
    }
  }
#pragma unroll
  for (int nt = 0; nt < 4; ++nt) {
    const int m = nt * 16 + l15;
#pragma unroll
    for (int r = 0; r < 8; ++r) {
      const int i = row0 + r + hsel * 8;  // global (b,s) row
      const int b = i / 100;
      const int s = i - b * 100;
      const float ph = __expf(acc[nt][r] * scale - 0.5f) * (1.0f / 64.0f);
      dst[((size_t)(b * 16 + h) * 100 + s) * 64 + m] = (_Float16)ph;
    }
  }
}

// ---------------- L-projections: phi_k_l[l,m] = sum_s phi_k*E; v_l[l,d]=sum_s v*F
__global__ __launch_bounds__(256) void k_lproj(
    const _Float16* __restrict__ phi_k16, const float* __restrict__ v,
    const float* __restrict__ E, const float* __restrict__ F,
    float* __restrict__ pkl, float* __restrict__ vl) {
  __shared__ float sP[100 * 64];
  __shared__ float sV[100 * 64];
  const int bh = blockIdx.x;
  const int b = bh >> 4, h = bh & 15;
  for (int idx = threadIdx.x; idx < 6400; idx += 256) {
    sP[idx] = (float)phi_k16[(size_t)bh * 6400 + idx];
    const int s = idx >> 6, d = idx & 63;
    sV[idx] = v[((size_t)(b * 100 + s)) * 1024 + h * 64 + d];
  }
  __syncthreads();
  for (int o = threadIdx.x; o < 48 * 128; o += 256) {
    const int l = o >> 7;
    const int c = o & 127;
    float acc = 0.f;
    if (c < 64) {
      for (int s = 0; s < 100; ++s) acc += sP[s * 64 + c] * E[s * 48 + l];
      pkl[((size_t)bh * 48 + l) * 64 + c] = acc;
    } else {
      const int d = c - 64;
      for (int s = 0; s < 100; ++s) acc += sV[s * 64 + d] * F[s * 48 + l];
      vl[((size_t)bh * 48 + l) * 64 + d] = acc;
    }
  }
}

// ---------------- A1t[d][m] = sum_l pkl[l][m] * vl[l][d]  (f16, B-ready) -----
__global__ __launch_bounds__(256) void k_a1(const float* __restrict__ pkl,
                                            const float* __restrict__ vl,
                                            _Float16* __restrict__ A1t) {
  __shared__ float sP[48 * 64];
  __shared__ float sV[48 * 64];
  const int bh = blockIdx.x;
  for (int idx = threadIdx.x; idx < 48 * 64; idx += 256) {
    sP[idx] = pkl[(size_t)bh * 3072 + idx];
    sV[idx] = vl[(size_t)bh * 3072 + idx];
  }
  __syncthreads();
  for (int o = threadIdx.x; o < 4096; o += 256) {
    const int d = o >> 6, m = o & 63;
    float acc = 0.f;
#pragma unroll 4
    for (int l = 0; l < 48; ++l) acc += sP[l * 64 + m] * sV[l * 64 + d];
    A1t[(size_t)bh * 4096 + d * 64 + m] = (_Float16)acc;
  }
}

// ---------------- num[s,d] = phi_q[s,:] @ A1  (WMMA, rows padded to 112) -----
// A1t tile (8KB) staged once per block in LDS via async copies.
__global__ __launch_bounds__(256) void k_num(const _Float16* __restrict__ phi_q16,
                                             const _Float16* __restrict__ A1t,
                                             float* __restrict__ numb) {
  __shared__ alignas(128) _Float16 sA1[64 * 64];  // 8KB
  const int bh = blockIdx.x;
  const int wave = threadIdx.x >> 5;
  const int lane = threadIdx.x & 31;
  const int hsel = lane >> 4;
  const int l15  = lane & 15;

  const _Float16* Bp = A1t + (size_t)bh * 4096;  // [d][m], m contiguous
  for (int i = threadIdx.x; i < 512; i += 256)   // 512 x 16B = 8KB
    async_ld16(Bp + i * 8, &sA1[i * 8]);
  wait_async0();
  __syncthreads();

  if (wave >= 7) return;  // 7*16 = 112 >= S=100 (after the cooperative copy)

  const int s0 = wave * 16;
  const _Float16* A = phi_q16 + (size_t)bh * 6400;  // [S,M], m contiguous

  const v8f vzero = {0.f,0.f,0.f,0.f,0.f,0.f,0.f,0.f};
  v8f acc[4];
#pragma unroll
  for (int i = 0; i < 4; ++i) acc[i] = vzero;

#pragma unroll
  for (int k0 = 0; k0 < 64; k0 += 32) {
    v16h a = load_a_frag(A, 64, s0 + l15, k0, hsel);  // pads past S, masked below
#pragma unroll
    for (int nt = 0; nt < 4; ++nt) {
      v16h bf = load_b_frag_lds(sA1, 64, nt * 16 + l15, k0, hsel);
      acc[nt] = wmma16(a, bf, acc[nt]);
    }
  }
#pragma unroll
  for (int nt = 0; nt < 4; ++nt) {
    const int d = nt * 16 + l15;
#pragma unroll
    for (int r = 0; r < 8; ++r) {
      const int s = s0 + r + hsel * 8;
      if (s < 100) numb[(size_t)bh * 6400 + s * 64 + d] = acc[nt][r];
    }
  }
}

// ---------------- den + divide + transpose to [B,S,H,D] ----------------------
__global__ __launch_bounds__(256) void k_final(
    const _Float16* __restrict__ phi_q16, const _Float16* __restrict__ phi_k16,
    const float* __restrict__ numb, float* __restrict__ out) {
  const int wave = threadIdx.x >> 5;
  const int lane = threadIdx.x & 31;
  const int gid = blockIdx.x * 8 + wave;  // [0, 2048*100)
  const int bh = gid / 100;
  const int s = gid - bh * 100;
  const int b = bh >> 4, h = bh & 15;
  const size_t base = (size_t)bh * 6400 + (size_t)s * 64;
  float den = (float)phi_q16[base + lane] * (float)phi_k16[base + lane] +
              (float)phi_q16[base + lane + 32] * (float)phi_k16[base + lane + 32];
#pragma unroll
  for (int off = 16; off > 0; off >>= 1) den += __shfl_xor(den, off, 32);
  const float rcp = 1.0f / den;
  const size_t ob = ((size_t)(b * 100 + s)) * 1024 + h * 64;
  out[ob + lane]      = numb[base + lane] * rcp;
  out[ob + lane + 32] = numb[base + lane + 32] * rcp;
}

extern "C" void kernel_launch(void* const* d_in, const int* in_sizes, int n_in,
                              void* d_out, int out_size, void* d_ws, size_t ws_size,
                              hipStream_t stream) {
  (void)in_sizes; (void)n_in; (void)out_size; (void)ws_size;
  const float* x   = (const float*)d_in[0];
  const float* Wq  = (const float*)d_in[1];
  const float* bq  = (const float*)d_in[2];
  const float* Wk  = (const float*)d_in[3];
  const float* bk  = (const float*)d_in[4];
  const float* Wv  = (const float*)d_in[5];
  const float* bvv = (const float*)d_in[6];
  const float* Wrf = (const float*)d_in[7];
  const float* E   = (const float*)d_in[8];
  const float* F   = (const float*)d_in[9];
  float* out = (float*)d_out;

  char* ws = (char*)d_ws;
  size_t off = 0;
  auto take = [&](size_t bytes) -> char* {
    char* p = ws + off;
    off += (bytes + 255) & ~(size_t)255;
    return p;
  };

  _Float16* xh   = (_Float16*)take((size_t)12800 * 1024 * 2);
  _Float16* Wh   = (_Float16*)take((size_t)3 * 1024 * 1024 * 2);
  _Float16* Wrfh = (_Float16*)take((size_t)16 * 64 * 64 * 2);
  _Float16* qh   = (_Float16*)take((size_t)12800 * 1024 * 2);
  _Float16* kh   = (_Float16*)take((size_t)12800 * 1024 * 2);
  float*    vbuf = (float*)take((size_t)12800 * 1024 * 4);
  _Float16* phiq = (_Float16*)take(((size_t)2048 * 6400 + 2048) * 2);  // +pad rows
  _Float16* phik = (_Float16*)take(((size_t)2048 * 6400 + 2048) * 2);
  float*    pkl  = (float*)take((size_t)2048 * 3072 * 4);
  float*    vl   = (float*)take((size_t)2048 * 3072 * 4);
  _Float16* A1t  = (_Float16*)take((size_t)2048 * 4096 * 2);
  float*    numb = (float*)take((size_t)2048 * 6400 * 4);
  float*    parts= (float*)take((size_t)2 * 12800 * 4);
  float*    inv  = (float*)take(256);

  // f32 -> f16 conversions
  k_f32tof16<<<(13107200 + 255) / 256, 256, 0, stream>>>(x, xh, 13107200);
  k_f32tof16<<<(1048576 + 255) / 256, 256, 0, stream>>>(Wq, Wh, 1048576);
  k_f32tof16<<<(1048576 + 255) / 256, 256, 0, stream>>>(Wk, Wh + 1048576, 1048576);
  k_f32tof16<<<(1048576 + 255) / 256, 256, 0, stream>>>(Wv, Wh + 2 * 1048576, 1048576);
  k_f32tof16<<<(65536 + 255) / 256, 256, 0, stream>>>(Wrf, Wrfh, 65536);

  // QKV projections (WMMA + async B staging) + norm partials
  k_gemm_qkv<<<dim3(100, 16, 3), 256, 0, stream>>>(xh, Wh, bq, bk, bvv,
                                                   qh, kh, vbuf, parts);
  // 1/||q||, 1/||k||
  k_norm<<<2, 256, 0, stream>>>(parts, inv);
  // random-feature maps (WMMA + exp epilogue, async Wrf staging)
  k_phi<<<dim3(100, 16, 2), 256, 0, stream>>>(qh, kh, Wrfh, inv, phiq, phik);
  // Linformer projections over sequence
  k_lproj<<<2048, 256, 0, stream>>>(phik, vbuf, E, F, pkl, vl);
  // A1^T (f16, B-fragment ready)
  k_a1<<<2048, 256, 0, stream>>>(pkl, vl, A1t);
  // numerator (WMMA, async A1 staging)
  k_num<<<2048, 256, 0, stream>>>(phiq, A1t, numb);
  // denominator + divide + transpose
  k_final<<<25600, 256, 0, stream>>>(phiq, phik, numb, out);
}
